// GraphPyramidModuleML_res_65867618451915
// MI455X (gfx1250) — matrix operand order repaired and checked by reading, not compiled, not run on hardware
//
#include <hip/hip_runtime.h>
#include <hip/hip_bf16.h>

// ---------------- Problem constants ----------------
#define NB   4
#define CCH  256
#define HWPX 16384            // 128*128
#define GH   512
#define HIDC 256
#define MAXK 20
#define EPSBN 1e-5f
#define OUT1SZ (NB * CCH * HWPX)   // 16,777,216 floats

typedef __attribute__((ext_vector_type(16))) _Float16 v16h;
typedef __attribute__((ext_vector_type(8)))  float    v8f;
typedef __attribute__((ext_vector_type(4)))  unsigned int tdm_g0_t;
typedef __attribute__((ext_vector_type(8)))  int          tdm_g1_t;
typedef __attribute__((ext_vector_type(4)))  int          tdm_g2_t;

__device__ __constant__ int c_lut2[20] =
    {0,1,1,3,1,2,2,2,4,4,2,2,3,1,2,2,4,4,4,4};

__device__ __forceinline__ int apply_lut(int lvl, int m) {
    return (lvl == 1) ? (m ? 1 : 0) : ((lvl == 2) ? c_lut2[m] : m);
}

// order-preserving encode for float atomicMax on unsigned
__device__ __forceinline__ unsigned ordenc(float f) {
    unsigned u = __float_as_uint(f);
    return (u & 0x80000000u) ? ~u : (u | 0x80000000u);
}
__device__ __forceinline__ float orddec(unsigned u) {
    u = (u & 0x80000000u) ? (u & 0x7FFFFFFFu) : ~u;
    return __uint_as_float(u);
}

// ---------------------------------------------------------------------
// Tensor Data Mover: 2D tile load (f32) global -> LDS.
// D# packed per CDNA5 ISA 8.3/8.4: group0 = {count/flags, lds_addr,
// global_addr lo, global_addr hi | type=2}; group1 = {data_size=4B,
// tensor_dim0/1, tile_dim0/1, dim0 stride}. Rows beyond tensor_dim1
// are zero-filled by TDM OOB semantics.
// ---------------------------------------------------------------------
__device__ __forceinline__ void tdm_load_2d(unsigned lds_off, const void* gptr,
                                            unsigned tile_d0, unsigned tile_d1,
                                            unsigned tensor_d0, unsigned tensor_d1,
                                            unsigned stride_d0)
{
    unsigned long long ga = (unsigned long long)(size_t)gptr;
    tdm_g0_t g0;
    g0[0] = 1u;                                   // count=1 (valid user D#)
    g0[1] = lds_off;                              // LDS byte address
    g0[2] = (unsigned)ga;                         // global_addr[31:0]
    g0[3] = (unsigned)((ga >> 32) & 0x1FFFFFFu) | (2u << 30);  // [56:32]|type=2
    tdm_g1_t g1;
    g1[0] = (int)(2u << 16);                      // data_size=2 (4 bytes)
    g1[1] = (int)((tensor_d0 & 0xFFFFu) << 16);   // tensor_dim0[15:0]
    g1[2] = (int)(((tensor_d0 >> 16) & 0xFFFFu) | ((tensor_d1 & 0xFFFFu) << 16));
    g1[3] = (int)(((tensor_d1 >> 16) & 0xFFFFu) | ((tile_d0 & 0xFFFFu) << 16));
    g1[4] = (int)(tile_d1 & 0xFFFFu);             // tile_dim1 (tile_dim2=0)
    g1[5] = (int)stride_d0;                       // tensor_dim0_stride[31:0]
    g1[6] = 0;
    g1[7] = 0;
    tdm_g2_t z4 = {0, 0, 0, 0};
#if defined(__clang_major__) && (__clang_major__ >= 23)
    tdm_g1_t z8 = {0, 0, 0, 0, 0, 0, 0, 0};
    __builtin_amdgcn_tensor_load_to_lds(g0, g1, z4, z4, z8, 0);
#else
    __builtin_amdgcn_tensor_load_to_lds(g0, g1, z4, z4, 0);
#endif
}

// =====================================================================
// Generic 1x1-conv GEMM via WMMA (Cin fixed = 256).
//   Y[n][co][p] = bias[co] + sum_c A(n,c,p) * W[co][c]  (+ resid[n][co][p])
// A source: direct  A = X[n][c][p]                 (G == nullptr)
//           gather  A = G[n][c][ lut(mask[n][p]) ] (G != nullptr)
// One wave per (16-pixel, 16-outch) tile; all 4 waves of a block share
// the same outch tile, whose weights are DMA'd to LDS by the TDM.
// =====================================================================
__global__ __launch_bounds__(128)
void conv_wmma_kernel(const float* __restrict__ X,
                      const float* __restrict__ G,
                      const int*   __restrict__ rawmask,
                      int lvl, int Knodes,
                      const float* __restrict__ W,
                      const float* __restrict__ bias,
                      const float* __restrict__ resid,
                      float* __restrict__ Y,
                      int Cout, int coshift)
{
    __shared__ __align__(16) float s_b[16 * CCH];   // 16 KB TDM-staged weights
    const int lane = threadIdx.x & 31;
    const int wid  = threadIdx.x >> 5;

    int w   = blockIdx.x * 4 + wid;                 // global wave id
    int n   = w >> (10 + coshift);                  // 1024 pixel tiles / batch
    int rem = w & ((1 << (10 + coshift)) - 1);
    int pt  = rem & 1023;
    int ct  = rem >> 10;                            // same for all 4 waves
    int p0  = pt << 4;
    int co0 = ct << 4;

    // ---- TDM: stage the 16-row weight tile into LDS (wave 0 issues) ----
    if (wid == 0) {
        tdm_load_2d((unsigned)(size_t)s_b, W + (size_t)co0 * CCH,
                    /*tile_d0=*/CCH, /*tile_d1=*/16,
                    /*tensor_d0=*/CCH, /*tensor_d1=*/(unsigned)(Cout - co0),
                    /*stride_d0=*/CCH);
        __builtin_amdgcn_s_wait_tensorcnt(0);
    }
    __syncthreads();

    const int p     = p0 + (lane & 15);       // this lane's pixel row (A: M)
    const int kbase = (lane >= 16) ? 16 : 0;  // A K-subgroup per lane group
    const int cb    = (lane < 16) ? 0 : 16;   // B K-subgroup per lane group

    int gidx = 0;
    if (G) gidx = apply_lut(lvl, rawmask[n * HWPX + p]);

    const float* Xn = X + (size_t)n * CCH * HWPX;
    const float* Gn = G ? (G + (size_t)n * CCH * Knodes) : X;

    const int coL = co0 + (lane & 15);        // this lane's out channel (B: N)
    const int coC = (coL < Cout) ? coL : (Cout - 1);
    const float* brow = s_b + (lane & 15) * CCH + cb;

    v8f acc = {};
#pragma unroll 2
    for (int c0 = 0; c0 < CCH; c0 += 32) {
        if (c0 + 32 < CCH) {  // warm next A chunk (gfx1250 global_prefetch_b8)
            if (!G) __builtin_prefetch(Xn + (size_t)(c0 + 32) * HWPX + p, 0, 1);
        }
        // ---- A tile (16x32 f16): contiguous k = 16*(lane/16) + h, same as B ----
        v16h a;
#pragma unroll
        for (int h = 0; h < 16; ++h) {
            int kk = h + kbase;
            int c  = c0 + kk;
            float v = G ? Gn[(size_t)c * Knodes + gidx]
                        : Xn[(size_t)c * HWPX + p];
            a[h] = (_Float16)v;
        }
        // ---- B tile (32x16 f16) from LDS: lane = N, halves = 16 K ----
        v16h b;
#pragma unroll
        for (int h = 0; h < 16; h += 4) {
            float4 f = *(const float4*)(brow + c0 + h);
            b[h + 0] = (_Float16)f.x;
            b[h + 1] = (_Float16)f.y;
            b[h + 2] = (_Float16)f.z;
            b[h + 3] = (_Float16)f.w;
        }
        acc = __builtin_amdgcn_wmma_f32_16x16x32_f16(
                  false, a, false, b, (short)0, acc, false, false);
    }

    if (coL < Cout) {
        float badd = bias[coC];
#pragma unroll
        for (int r = 0; r < 8; ++r) {
            size_t idx = (size_t)n * Cout * HWPX + (size_t)coL * HWPX
                       + p0 + r + ((lane >> 4) << 3);
            float v = acc[r] + badd;
            if (resid) v += resid[idx];
            Y[idx] = v;
        }
    }
}

// =====================================================================
// argmax over 20 sem channels -> raw mask
// =====================================================================
__global__ void argmax_kernel(const float* __restrict__ xaux,
                              int* __restrict__ rawmask)
{
    int p = blockIdx.x * 256 + threadIdx.x;
    if (p >= NB * HWPX) return;
    int n = p / HWPX, pp = p % HWPX;
    const float* src = xaux + (size_t)n * 20 * HWPX + pp;
    float best = src[0];
    int bi = 0;
    for (int k = 1; k < 20; ++k) {
        float v = src[(size_t)k * HWPX];
        if (v > best) { best = v; bi = k; }
    }
    rawmask[p] = bi;
}

// =====================================================================
// graph_semantic_aggregation: per (n, channel) block computes segment
// mean & max(0-baseline) over pixels -> g[n][k][0:256]=ave, [256:512]=max
// =====================================================================
__global__ __launch_bounds__(256)
void agg_kernel(const float* __restrict__ X, const int* __restrict__ rawmask,
                int lvl, int K, float* __restrict__ g)
{
    int n = blockIdx.x >> 8;
    int c = blockIdx.x & 255;
    __shared__ float    s_sum[MAXK];
    __shared__ unsigned s_maxu[MAXK];
    __shared__ unsigned s_cnt[MAXK];
    if (threadIdx.x < K) {
        s_sum[threadIdx.x]  = 0.f;
        s_maxu[threadIdx.x] = ordenc(0.0f);   // onehot product includes 0
        s_cnt[threadIdx.x]  = 0u;
    }
    __syncthreads();
    const float* Xc   = X + ((size_t)n * CCH + c) * HWPX;
    const int*   mrow = rawmask + n * HWPX;
    for (int p = threadIdx.x; p < HWPX; p += 256) {
        int k = apply_lut(lvl, mrow[p]);
        float v = Xc[p];
        atomicAdd(&s_sum[k], v);
        atomicMax(&s_maxu[k], ordenc(v));
        atomicAdd(&s_cnt[k], 1u);
    }
    __syncthreads();
    if (threadIdx.x < K) {
        int k = threadIdx.x;
        float cnt = (float)s_cnt[k];
        float den = (cnt == 0.f) ? 1.f : cnt;
        size_t base = ((size_t)n * K + k) * (2 * CCH);
        g[base + c]       = s_sum[k] / den;
        g[base + CCH + c] = orddec(s_maxu[k]);
    }
}

// =====================================================================
// graph transform: xg[n][o][k] = gt_b[o] + sum_c g[n][k][c] * gt_w[o][c]
// =====================================================================
__global__ __launch_bounds__(256)
void gt_kernel(const float* __restrict__ g, const float* __restrict__ gw,
               const float* __restrict__ gb, int K, float* __restrict__ xg)
{
    int n = blockIdx.x;
    __shared__ float s_g[MAXK * GH];   // 40 KB
    for (int i = threadIdx.x; i < K * GH; i += 256)
        s_g[i] = g[(size_t)n * K * GH + i];
    __syncthreads();
    for (int o = threadIdx.x; o < GH; o += 256) {
        const float* wrow = gw + (size_t)o * GH;
        for (int k = 0; k < K; ++k) {
            float s = gb[o];
            const float* gk = s_g + k * GH;
            for (int c = 0; c < GH; ++c) s += gk[c] * wrow[c];
            xg[((size_t)n * GH + o) * K + k] = s;
        }
    }
}

// =====================================================================
// BatchNorm1d statistics over (batch, nodes) — separate kernel so the
// in-place attention update cannot race the cross-batch reduction.
// =====================================================================
__global__ void bn_stats_kernel(const float* __restrict__ xg, int K,
                                float* __restrict__ stats)
{
    int o = blockIdx.x * 256 + threadIdx.x;
    if (o >= GH) return;
    float s = 0.f, s2 = 0.f;
    for (int nn = 0; nn < NB; ++nn) {
        const float* row = xg + ((size_t)nn * GH + o) * K;
        for (int k = 0; k < K; ++k) { float v = row[k]; s += v; s2 += v * v; }
    }
    float inv = 1.0f / (float)(NB * K);
    float mu  = s * inv;
    float var = s2 * inv - mu * mu;
    stats[o]      = mu;
    stats[GH + o] = rsqrtf(var + EPSBN);
}

// =====================================================================
// BN-apply + SAGAN self-attention over nodes + ReLU, in-place on xg.
// One block per batch sample; everything lives in LDS (~52 KB).
// =====================================================================
__global__ __launch_bounds__(256)
void attn_kernel(float* __restrict__ xg, int K,
                 const float* __restrict__ stats,
                 const float* __restrict__ bng, const float* __restrict__ bnb,
                 const float* __restrict__ wq, const float* __restrict__ bq,
                 const float* __restrict__ wk, const float* __restrict__ bk,
                 const float* __restrict__ wv, const float* __restrict__ bv,
                 const float* __restrict__ gamma)
{
    int n = blockIdx.x;
    __shared__ float s_x[GH * MAXK];          // 40 KB, BN'd x for this n
    __shared__ float s_q[64 * MAXK];          // 5 KB
    __shared__ float s_k[64 * MAXK];          // 5 KB
    __shared__ float s_att[MAXK * MAXK];

    for (int i = threadIdx.x; i < GH * K; i += 256) {
        int o = i / K, k = i % K;
        float v = xg[((size_t)n * GH + o) * K + k];
        s_x[o * K + k] = (v - stats[o]) * stats[GH + o] * bng[o] + bnb[o];
    }
    __syncthreads();

    for (int i = threadIdx.x; i < 64 * K; i += 256) {
        int o = i / K, k = i % K;
        const float* wr = wq + (size_t)o * GH;
        float s = bq[o];
        for (int c = 0; c < GH; ++c) s += s_x[c * K + k] * wr[c];
        s_q[o * K + k] = s;
        wr = wk + (size_t)o * GH;
        s = bk[o];
        for (int c = 0; c < GH; ++c) s += s_x[c * K + k] * wr[c];
        s_k[o * K + k] = s;
    }
    __syncthreads();

    for (int t = threadIdx.x; t < K * K; t += 256) {
        int i = t / K, j = t % K;
        float s = 0.f;
        for (int o = 0; o < 64; ++o) s += s_q[o * K + i] * s_k[o * K + j];
        s_att[i * K + j] = s;
    }
    __syncthreads();

    if (threadIdx.x < K) {
        int i = threadIdx.x;
        float m = -1e30f;
        for (int j = 0; j < K; ++j) m = fmaxf(m, s_att[i * K + j]);
        float sum = 0.f;
        for (int j = 0; j < K; ++j) {
            float e = __expf(s_att[i * K + j] - m);
            s_att[i * K + j] = e;
            sum += e;
        }
        float r = 1.0f / sum;
        for (int j = 0; j < K; ++j) s_att[i * K + j] *= r;
    }
    __syncthreads();

    float gm = gamma[0];
    for (int c = threadIdx.x; c < GH; c += 256) {
        float vrow[MAXK];
        const float* wr = wv + (size_t)c * GH;
#pragma unroll
        for (int j = 0; j < MAXK; ++j) vrow[j] = 0.f;
#pragma unroll
        for (int j = 0; j < MAXK; ++j) {
            if (j < K) {
                float s = bv[c];
                for (int cc = 0; cc < GH; ++cc) s += s_x[cc * K + j] * wr[cc];
                vrow[j] = s;
            }
        }
#pragma unroll
        for (int i = 0; i < MAXK; ++i) {
            if (i < K) {
                float s = 0.f;
#pragma unroll
                for (int j = 0; j < MAXK; ++j)
                    if (j < K) s += vrow[j] * s_att[i * K + j];
                float x = gm * s + s_x[c * K + i];
                xg[((size_t)n * GH + c) * K + i] = fmaxf(x, 0.f);  // relu
            }
        }
    }
}

// =====================================================================
// back transform: gl[n][o][k] = bb[o] + sum_c xg[n][c][k]*bw[o][c], o<256
// =====================================================================
__global__ __launch_bounds__(256)
void back_kernel(const float* __restrict__ xg, const float* __restrict__ bw,
                 const float* __restrict__ bb, int K, float* __restrict__ gl)
{
    int n = blockIdx.x;
    __shared__ float s_x[GH * MAXK];   // 40 KB
    for (int i = threadIdx.x; i < GH * K; i += 256)
        s_x[i] = xg[(size_t)n * GH * K + i];
    __syncthreads();
    for (int o = threadIdx.x; o < HIDC; o += 256) {
        const float* wrow = bw + (size_t)o * GH;
        for (int k = 0; k < K; ++k) {
            float s = bb[o];
            for (int c = 0; c < GH; ++c) s += s_x[c * K + k] * wrow[c];
            gl[((size_t)n * HIDC + o) * K + k] = s;
        }
    }
}

// =====================================================================
// Final fusion: cat(features,fused1,fused2,fused3)[1024] -> 512 -> relu(256)
// Fused through LDS: GEMM-1 writes its 16x512 hidden tile in WMMA A-layout
// f16 into LDS; GEMM-2 reads it with contiguous 32B ds reads. Never
// materializes the 128 MB hidden tensor in HBM. One wave per 16 pixels.
// =====================================================================
__global__ __launch_bounds__(32)
void fusion_kernel(const float* __restrict__ feats,
                   const float* __restrict__ f1,
                   const float* __restrict__ f2,
                   const float* __restrict__ f3,
                   const float* __restrict__ w1, const float* __restrict__ b1,
                   const float* __restrict__ w2, const float* __restrict__ b2,
                   float* __restrict__ out)
{
    __shared__ __align__(32) _Float16 s_h[16 * 32 * 16];  // 16 K-chunks, 16 KB
    const int lane = threadIdx.x;
    int w  = blockIdx.x;                  // over NB*(HWPX/16)
    int n  = w >> 10;
    int p0 = (w & 1023) << 4;

    const float* s0 = feats + (size_t)n * CCH * HWPX;
    const float* s1 = f1    + (size_t)n * CCH * HWPX;
    const float* s2 = f2    + (size_t)n * CCH * HWPX;
    const float* s3 = f3    + (size_t)n * CCH * HWPX;

    const int kbase = (lane >= 16) ? 16 : 0;
    const int pL    = p0 + (lane & 15);
    const int cb    = (lane < 16) ? 0 : 16;

    // -------- GEMM 1: [16,1024] x [1024,512] --------
    for (int ctile = 0; ctile < 32; ++ctile) {
        int co0 = ctile * 16;
        v8f acc = {};
        for (int kc = 0; kc < 32; ++kc) {
            int cbase = kc * 32;
            const float* S = (kc < 8) ? s0 : (kc < 16) ? s1 : (kc < 24) ? s2 : s3;
            int crel = cbase & 255;
            v16h a;
#pragma unroll
            for (int h = 0; h < 16; ++h) {
                int kk = h + kbase;
                a[h] = (_Float16)S[(size_t)(crel + kk) * HWPX + pL];
            }
            const float* wrow = w1 + (size_t)(co0 + (lane & 15)) * 1024 + cbase + cb;
            v16h b;
#pragma unroll
            for (int h = 0; h < 16; h += 4) {
                float4 f = *(const float4*)(wrow + h);
                b[h + 0] = (_Float16)f.x; b[h + 1] = (_Float16)f.y;
                b[h + 2] = (_Float16)f.z; b[h + 3] = (_Float16)f.w;
            }
            acc = __builtin_amdgcn_wmma_f32_16x16x32_f16(
                      false, a, false, b, (short)0, acc, false, false);
        }
        // scatter D tile into A-layout LDS for GEMM 2
        int co    = co0 + (lane & 15);
        float bb  = b1[co];
        int chunk = co >> 5;
        int cr    = co & 31;
        int lgrp  = (cr & 16) ? 16 : 0;
        int h2    = cr & 15;
#pragma unroll
        for (int r = 0; r < 8; ++r) {
            int m = r + ((lane >> 4) << 3);
            s_h[(chunk * 32 + m + lgrp) * 16 + h2] = (_Float16)(acc[r] + bb);
        }
    }
    __syncthreads();

    // -------- GEMM 2: [16,512] x [512,256] + relu + residual --------
    for (int ot = 0; ot < 16; ++ot) {
        v8f acc = {};
        for (int kc = 0; kc < 16; ++kc) {
            v16h a = *(const v16h*)(s_h + (kc * 32 + lane) * 16);
            const float* wrow = w2 + (size_t)(ot * 16 + (lane & 15)) * 512
                              + kc * 32 + cb;
            v16h b;
#pragma unroll
            for (int h = 0; h < 16; h += 4) {
                float4 f = *(const float4*)(wrow + h);
                b[h + 0] = (_Float16)f.x; b[h + 1] = (_Float16)f.y;
                b[h + 2] = (_Float16)f.z; b[h + 3] = (_Float16)f.w;
            }
            acc = __builtin_amdgcn_wmma_f32_16x16x32_f16(
                      false, a, false, b, (short)0, acc, false, false);
        }
        int co = ot * 16 + (lane & 15);
        float bb = b2[co];
#pragma unroll
        for (int r = 0; r < 8; ++r) {
            size_t idx = (size_t)n * CCH * HWPX + (size_t)co * HWPX
                       + p0 + r + ((lane >> 4) << 3);
            float v = fmaxf(acc[r] + bb, 0.f);
            out[idx] = feats[idx] + v;
        }
    }
}

// =====================================================================
// Host-side orchestration
// =====================================================================
extern "C" void kernel_launch(void* const* d_in, const int* in_sizes, int n_in,
                              void* d_out, int out_size, void* d_ws, size_t ws_size,
                              hipStream_t stream)
{
    // Flattened pytree, insertion order:
    // 0:features 1:sem_w 2:sem_b
    // blk1: 3..33  blk2: 34..64  blk3: 65..95   (31 leaves each:
    //   +0 gt_w +1 gt_b +2 back_w +3 back_b, then per i: +4+9i bn_g,
    //   +5+9i bn_b, +6+9i wq,bq,wk,bk,wv,bv,gamma)
    // 96:t1_w 97:t1_b 98:t2_w 99:t2_b 100:t3_w 101:t3_b
    // 102:fus1_w 103:fus1_b 104:fus2_w 105:fus2_b 106:dataset
    const float* feats = (const float*)d_in[0];
    const float* sem_w = (const float*)d_in[1];
    const float* sem_b = (const float*)d_in[2];
    const int blkBase[3] = {3, 34, 65};
    const float* t_w[3] = {(const float*)d_in[96], (const float*)d_in[98],
                           (const float*)d_in[100]};
    const float* t_b[3] = {(const float*)d_in[97], (const float*)d_in[99],
                           (const float*)d_in[101]};

    // workspace layout (256B aligned)
    char* ws = (char*)d_ws;
    size_t off = 0;
    auto take = [&](size_t bytes) {
        char* p = ws + off;
        off = (off + bytes + 255) & ~(size_t)255;
        return p;
    };
    int*   rawmask = (int*)  take((size_t)NB * HWPX * 4);
    float* g_in    = (float*)take((size_t)NB * MAXK * 2 * CCH * 4);
    float* xg      = (float*)take((size_t)NB * GH * MAXK * 4);
    float* gl      = (float*)take((size_t)NB * HIDC * MAXK * 4);
    float* stats   = (float*)take((size_t)2 * GH * 4);
    float* fused[3];
    for (int i = 0; i < 3; ++i)
        fused[i] = (float*)take((size_t)NB * CCH * HWPX * 4);

    float* out1 = (float*)d_out;
    float* xaux = (float*)d_out + OUT1SZ;

    // 1) semantic head (256 -> 20) + argmax
    {
        int coshift = 1;  // 2 co-tiles: 20 out channels padded to 32
        int waves   = NB * 1024 * 2;
        conv_wmma_kernel<<<waves / 4, 128, 0, stream>>>(
            feats, nullptr, rawmask, 0, 1, sem_w, sem_b, nullptr, xaux,
            20, coshift);
    }
    argmax_kernel<<<(NB * HWPX + 255) / 256, 256, 0, stream>>>(xaux, rawmask);

    // 2) three pyramid levels
    const float* src = feats;
    const int Ks[3]  = {2, 5, 20};
    for (int L = 0; L < 3; ++L) {
        int base = blkBase[L];
        int K    = Ks[L];
        int lvl  = L + 1;

        agg_kernel<<<NB * 256, 256, 0, stream>>>(src, rawmask, lvl, K, g_in);
        gt_kernel<<<NB, 256, 0, stream>>>(
            g_in, (const float*)d_in[base + 0], (const float*)d_in[base + 1],
            K, xg);
        for (int i = 0; i < 3; ++i) {
            int ab = base + 6 + 9 * i;
            bn_stats_kernel<<<2, 256, 0, stream>>>(xg, K, stats);
            attn_kernel<<<NB, 256, 0, stream>>>(
                xg, K, stats,
                (const float*)d_in[base + 4 + 9 * i],
                (const float*)d_in[base + 5 + 9 * i],
                (const float*)d_in[ab + 0], (const float*)d_in[ab + 1],
                (const float*)d_in[ab + 2], (const float*)d_in[ab + 3],
                (const float*)d_in[ab + 4], (const float*)d_in[ab + 5],
                (const float*)d_in[ab + 6]);
        }
        back_kernel<<<NB, 256, 0, stream>>>(
            xg, (const float*)d_in[base + 2], (const float*)d_in[base + 3],
            K, gl);

        // fusedL = src + conv1x1(gather(gl, mask_L), t_w[L])
        int waves = NB * 1024 * 16;   // 16 co-tiles
        conv_wmma_kernel<<<waves / 4, 128, 0, stream>>>(
            src, gl, rawmask, lvl, K, t_w[L], t_b[L], src, fused[L],
            256, 4);
        src = fused[L];
    }

    // 3) final fusion (1024 -> 512 -> relu 256) + residual, fused in LDS
    fusion_kernel<<<NB * 1024, 32, 0, stream>>>(
        feats, fused[0], fused[1], fused[2],
        (const float*)d_in[102], (const float*)d_in[103],
        (const float*)d_in[104], (const float*)d_in[105], out1);

    (void)in_sizes; (void)n_in; (void)out_size; (void)ws_size;
}
